// BidirectionalNet_15479062135021
// MI455X (gfx1250) — matrix operand order repaired and checked by reading, not compiled, not run on hardware
//
#include <hip/hip_runtime.h>
#include <math.h>

#define NN 100000   // nodes
#define NE 400000   // edges
#define NF 128      // feature dim entering every conv (layer1: 128, layer2: 128)
#define NG 512      // graphs
#define NC 8        // classes

typedef __attribute__((ext_vector_type(16))) __bf16 v16bf;
typedef __attribute__((ext_vector_type(8)))  float  v8f;

// ---------------------------------------------------------------- utilities

__global__ __launch_bounds__(256) void k_zero4(float4* p, int n4) {
    int i = blockIdx.x * blockDim.x + threadIdx.x;
    if (i < n4) p[i] = make_float4(0.f, 0.f, 0.f, 0.f);
}

__global__ __launch_bounds__(256) void k_negcopy4(const float4* in, float4* out, int n4) {
    int i = blockIdx.x * blockDim.x + threadIdx.x;
    if (i < n4) {
        float4 v = in[i];
        out[i] = make_float4(-v.x, -v.y, -v.z, -v.w);
    }
}

// ---------------------------------------------------------------- norms

__global__ __launch_bounds__(256) void k_edge_deg(const int* a, const int* b, float* deg, int nE) {
    int e = blockIdx.x * blockDim.x + threadIdx.x;
    if (e < nE) {
        int s = a[e], d = b[e];
        if (s != d) atomicAdd(&deg[s], 1.0f);
    }
}

__global__ __launch_bounds__(256) void k_dis(float* deg, int n) {
    int i = blockIdx.x * blockDim.x + threadIdx.x;
    if (i < n) {
        float d = deg[i];
        deg[i] = d > 0.f ? rsqrtf(d) : 0.f;
    }
}

__global__ __launch_bounds__(256) void k_norm(const int* a, const int* b, const float* dis,
                                              float* nrm, int nE) {
    int e = blockIdx.x * blockDim.x + threadIdx.x;
    if (e < nE) {
        int s = a[e], d = b[e];
        nrm[e] = (s != d) ? (-dis[s] * dis[d]) : 0.f;
    }
}

// ------------------------------------------------- sparse propagation (scatter)
// out[sidx[e], :] += scale * nrm[e] * t[gidx[e], :]   (128 feats, float4/thread)

__global__ __launch_bounds__(256) void k_prop(const float* __restrict__ t,
                                              const float* __restrict__ nrm,
                                              const int* __restrict__ gidx,
                                              const int* __restrict__ sidx,
                                              float* __restrict__ out,
                                              float scale, int nE) {
    int i = blockIdx.x * blockDim.x + threadIdx.x;
    int e = i >> 5;
    if (e >= nE) return;
    float s = scale * nrm[e];
    if (s == 0.f) return;                 // self-loops / isolated: no work
    int f = (i & 31) << 2;
    const float4 v = *(const float4*)(t + (size_t)gidx[e] * NF + f);
    float* o = out + (size_t)sidx[e] * NF + f;
    atomicAdd(o + 0, s * v.x);
    atomicAdd(o + 1, s * v.y);
    atomicAdd(o + 2, s * v.z);
    atomicAdd(o + 3, s * v.w);
}

// ---------------------------------------------------- weight pack to WMMA B layout
// B (32x16 KxN per step): elems 0..7 -> K = hi*8+0..7, elems 8..15 -> K = 16+hi*8+0..7 (hi = lane>=16).
// Bp[((kk*4+ks)*(OUT/16)+cg)*32 + lane][j] = bf16( W[kk][ks*32 + hi*8 + j + (j>=8)*8][cg*16 + lane%16] )

__global__ __launch_bounds__(256) void k_pack_w(const float* __restrict__ W,
                                                __bf16* __restrict__ Bp, int OUT) {
    int idx = blockIdx.x * blockDim.x + threadIdx.x;
    int total = 4 * 4 * (OUT / 16) * 32;
    if (idx >= total) return;
    int lane = idx & 31;
    int r = idx >> 5;
    int cg = r % (OUT / 16); r /= (OUT / 16);
    int ks = r & 3;
    int kk = r >> 2;
    int hi = (lane >> 4) & 1;
    int n = cg * 16 + (lane & 15);
    __bf16* dst = Bp + (size_t)idx * 16;
#pragma unroll
    for (int j = 0; j < 16; ++j) {
        int k = ks * 32 + hi * 8 + j + (j >= 8 ? 8 : 0);
        dst[j] = (__bf16)W[((size_t)kk * 128 + k) * OUT + n];
    }
}

// -------------------------------------------------------------- WMMA GEMM
// C[N x ldc](+= or =) A[N x 128] @ Wk[128 x OUT], one wave = 16 rows x 64 cols,
// K swept in four bf16 16x16x32 WMMA steps. A layout per ISA: lane m=lane&15,
// elems 0..7 -> K = kb+koff+0..7, elems 8..15 -> K = kb+16+koff+0..7, koff = (lane>=16)*8.

__global__ __launch_bounds__(256) void k_gemm_cheb(const float* __restrict__ A,
                                                   const __bf16* __restrict__ Bp,
                                                   float* __restrict__ C,
                                                   int ldc, int colOff, int nCol64,
                                                   int rowTiles, int beta) {
    int wid = (int)((blockIdx.x * blockDim.x + threadIdx.x) >> 5);
    int rowTile = wid / nCol64;
    if (rowTile >= rowTiles) return;          // wave-uniform: EXEC stays all-ones
    int cg64 = wid - rowTile * nCol64;
    int lane = threadIdx.x & 31;
    int hi = lane >> 4;
    int mrow = lane & 15;
    int rowBase = rowTile * 16;
    int colBase = colOff + cg64 * 64;

    v8f acc[4];
    if (beta) {
#pragma unroll
        for (int t = 0; t < 4; ++t) {
            int col = colBase + t * 16 + mrow;
#pragma unroll
            for (int v = 0; v < 8; ++v)
                acc[t][v] = C[(size_t)(rowBase + hi * 8 + v) * ldc + col];
        }
    } else {
#pragma unroll
        for (int t = 0; t < 4; ++t)
#pragma unroll
            for (int v = 0; v < 8; ++v) acc[t][v] = 0.f;
    }

    const float* arow = A + (size_t)(rowBase + mrow) * NF + hi * 8;
    int nCg16 = nCol64 * 4;
#pragma unroll
    for (int ks = 0; ks < 4; ++ks) {
        const float* ap = arow + ks * 32;
        float4 f0 = *(const float4*)(ap);
        float4 f1 = *(const float4*)(ap + 4);
        float4 f2 = *(const float4*)(ap + 16);
        float4 f3 = *(const float4*)(ap + 20);
        v16bf a;
        a[0]  = (__bf16)f0.x; a[1]  = (__bf16)f0.y; a[2]  = (__bf16)f0.z; a[3]  = (__bf16)f0.w;
        a[4]  = (__bf16)f1.x; a[5]  = (__bf16)f1.y; a[6]  = (__bf16)f1.z; a[7]  = (__bf16)f1.w;
        a[8]  = (__bf16)f2.x; a[9]  = (__bf16)f2.y; a[10] = (__bf16)f2.z; a[11] = (__bf16)f2.w;
        a[12] = (__bf16)f3.x; a[13] = (__bf16)f3.y; a[14] = (__bf16)f3.z; a[15] = (__bf16)f3.w;
#pragma unroll
        for (int t = 0; t < 4; ++t) {
            v16bf b = *(const v16bf*)(Bp + ((size_t)(ks * nCg16 + cg64 * 4 + t) * 32 + lane) * 16);
            acc[t] = __builtin_amdgcn_wmma_f32_16x16x32_bf16(
                false, a, false, b, (short)0, acc[t], false, false);
        }
    }

#pragma unroll
    for (int t = 0; t < 4; ++t) {
        int col = colBase + t * 16 + mrow;
#pragma unroll
        for (int v = 0; v < 8; ++v)
            C[(size_t)(rowBase + hi * 8 + v) * ldc + col] = acc[t][v];
    }
}

// -------------------------------------------------------------- epilogues

__global__ __launch_bounds__(256) void k_bias_relu(float* C, const float* bias,
                                                   int ldc, int colOff, int ncols, int nrows) {
    int i = blockIdx.x * blockDim.x + threadIdx.x;
    int row = i / ncols;
    int col = i - row * ncols;
    if (row >= nrows) return;
    float* p = C + (size_t)row * ldc + colOff + col;
    float v = *p + bias[col];
    *p = v > 0.f ? v : 0.f;
}

__global__ __launch_bounds__(256) void k_pool_add(const float* __restrict__ H,
                                                  const int* __restrict__ batch,
                                                  float* __restrict__ pooled,
                                                  int ncols, int colOff, int nrows) {
    int i = blockIdx.x * blockDim.x + threadIdx.x;
    int c4cnt = ncols >> 2;
    int row = i / c4cnt;
    if (row >= nrows) return;
    int c4 = (i - row * c4cnt) << 2;
    int g = batch[row];
    const float4 v = *(const float4*)(H + (size_t)row * ncols + c4);
    float* o = pooled + (size_t)g * 512 + colOff + c4;
    atomicAdd(o + 0, v.x);
    atomicAdd(o + 1, v.y);
    atomicAdd(o + 2, v.z);
    atomicAdd(o + 3, v.w);
}

__global__ __launch_bounds__(256) void k_count(const int* batch, float* cnt, int n) {
    int i = blockIdx.x * blockDim.x + threadIdx.x;
    if (i < n) atomicAdd(&cnt[batch[i]], 1.0f);
}

__global__ __launch_bounds__(256) void k_fc_lsm(const float* __restrict__ pooled,
                                                const float* __restrict__ cnt,
                                                const float* __restrict__ fw,
                                                const float* __restrict__ fb,
                                                float* __restrict__ out) {
    int g = blockIdx.x * blockDim.x + threadIdx.x;
    if (g >= NG) return;
    float inv = 1.0f / fmaxf(cnt[g], 1.0f);
    float logit[NC];
#pragma unroll
    for (int c = 0; c < NC; ++c) logit[c] = fb[c];
    const float* pr = pooled + (size_t)g * 512;
    for (int j = 0; j < 512; ++j) {
        float pv = pr[j] * inv;
#pragma unroll
        for (int c = 0; c < NC; ++c) logit[c] += pv * fw[j * NC + c];
    }
    float m = logit[0];
#pragma unroll
    for (int c = 1; c < NC; ++c) m = fmaxf(m, logit[c]);
    float s = 0.f;
#pragma unroll
    for (int c = 0; c < NC; ++c) s += expf(logit[c] - m);
    float ls = logf(s);
#pragma unroll
    for (int c = 0; c < NC; ++c) out[(size_t)g * NC + c] = logit[c] - m - ls;
}

// -------------------------------------------------------- one ChebConv (K=4)

static void run_conv(hipStream_t stream, const float* X, const float* nrm,
                     const int* gidx, const int* sidx, const __bf16* Wp, int OUT,
                     float* C, int ldc, int colOff, float* B1, float* B2) {
    const int rowTiles = NN / 16;            // 6250
    const int nCol64 = OUT / 64;
    const int gblocks = (rowTiles * nCol64 * 32 + 255) / 256;
    const size_t wstride = (size_t)128 * OUT;  // bf16 elems per cheb order
    const int n4 = NN * NF / 4;
    const int zblocks = (n4 + 255) / 256;
    const int pblocks = (NE * 32 + 255) / 256;

    // k=0: Tx0 = X
    k_gemm_cheb<<<gblocks, 256, 0, stream>>>(X, Wp, C, ldc, colOff, nCol64, rowTiles, 0);
    // Tx1 = L_hat X
    k_zero4<<<zblocks, 256, 0, stream>>>((float4*)B1, n4);
    k_prop<<<pblocks, 256, 0, stream>>>(X, nrm, gidx, sidx, B1, 1.0f, NE);
    k_gemm_cheb<<<gblocks, 256, 0, stream>>>(B1, Wp + wstride, C, ldc, colOff, nCol64, rowTiles, 1);
    // Tx2 = 2 L_hat Tx1 - Tx0  (init target to -Tx0, scatter with scale 2)
    k_negcopy4<<<zblocks, 256, 0, stream>>>((const float4*)X, (float4*)B2, n4);
    k_prop<<<pblocks, 256, 0, stream>>>(B1, nrm, gidx, sidx, B2, 2.0f, NE);
    k_gemm_cheb<<<gblocks, 256, 0, stream>>>(B2, Wp + 2 * wstride, C, ldc, colOff, nCol64, rowTiles, 1);
    // Tx3 = 2 L_hat Tx2 - Tx1  (negate B1 in place, scatter with scale 2)
    k_negcopy4<<<zblocks, 256, 0, stream>>>((const float4*)B1, (float4*)B1, n4);
    k_prop<<<pblocks, 256, 0, stream>>>(B2, nrm, gidx, sidx, B1, 2.0f, NE);
    k_gemm_cheb<<<gblocks, 256, 0, stream>>>(B1, Wp + 3 * wstride, C, ldc, colOff, nCol64, rowTiles, 1);
}

// ------------------------------------------------------------------ launcher

extern "C" void kernel_launch(void* const* d_in, const int* in_sizes, int n_in,
                              void* d_out, int out_size, void* d_ws, size_t ws_size,
                              hipStream_t stream) {
    const float* x   = (const float*)d_in[0];
    const int*   ei  = (const int*)d_in[1];    // [2, NE], integer -> const int* per harness
    const int* batch = (const int*)d_in[2];
    const float* W11 = (const float*)d_in[3];
    const float* b11 = (const float*)d_in[4];
    const float* W12 = (const float*)d_in[5];
    const float* b12 = (const float*)d_in[6];
    const float* W21 = (const float*)d_in[7];
    const float* b21 = (const float*)d_in[8];
    const float* W22 = (const float*)d_in[9];
    const float* b22 = (const float*)d_in[10];
    const float* fcw = (const float*)d_in[11];
    const float* fcb = (const float*)d_in[12];
    float* out = (float*)d_out;

    const int* src = ei;
    const int* dst = ei + NE;

    // workspace carve-out (256B aligned)
    char* ws = (char*)d_ws;
    auto take = [&](size_t bytes) -> char* {
        char* p = ws;
        ws += (bytes + 255) & ~(size_t)255;
        return p;
    };
    float*  norm_f = (float*)take((size_t)NE * 4);
    float*  norm_r = (float*)take((size_t)NE * 4);
    float*  deg    = (float*)take((size_t)NN * 4);
    float*  cnt    = (float*)take((size_t)NG * 4);
    float*  pooled = (float*)take((size_t)NG * 512 * 4);
    float*  B1     = (float*)take((size_t)NN * NF * 4);
    float*  B2     = (float*)take((size_t)NN * NF * 4);
    float*  h1     = (float*)take((size_t)NN * NF * 4);
    float*  out2   = (float*)take((size_t)NN * 256 * 4);
    __bf16* p11    = (__bf16*)take((size_t)4 * 128 * 64 * 2);
    __bf16* p12    = (__bf16*)take((size_t)4 * 128 * 64 * 2);
    __bf16* p21    = (__bf16*)take((size_t)4 * 128 * 256 * 2);
    __bf16* p22    = (__bf16*)take((size_t)4 * 128 * 256 * 2);

    auto nb = [](int n) { return (n + 255) / 256; };

    // pack weights into WMMA B layout (bf16)
    k_pack_w<<<nb(4 * 4 * 4 * 32), 256, 0, stream>>>(W11, p11, 64);
    k_pack_w<<<nb(4 * 4 * 4 * 32), 256, 0, stream>>>(W12, p12, 64);
    k_pack_w<<<nb(4 * 4 * 16 * 32), 256, 0, stream>>>(W21, p21, 256);
    k_pack_w<<<nb(4 * 4 * 16 * 32), 256, 0, stream>>>(W22, p22, 256);

    // edge normalizations (forward / reverse)
    k_zero4<<<nb(NN / 4), 256, 0, stream>>>((float4*)deg, NN / 4);
    k_edge_deg<<<nb(NE), 256, 0, stream>>>(src, dst, deg, NE);
    k_dis<<<nb(NN), 256, 0, stream>>>(deg, NN);
    k_norm<<<nb(NE), 256, 0, stream>>>(src, dst, deg, norm_f, NE);
    k_zero4<<<nb(NN / 4), 256, 0, stream>>>((float4*)deg, NN / 4);
    k_edge_deg<<<nb(NE), 256, 0, stream>>>(dst, src, deg, NE);
    k_dis<<<nb(NN), 256, 0, stream>>>(deg, NN);
    k_norm<<<nb(NE), 256, 0, stream>>>(dst, src, deg, norm_r, NE);

    // pooling state
    k_zero4<<<nb(NG / 4), 256, 0, stream>>>((float4*)cnt, NG / 4);
    k_count<<<nb(NN), 256, 0, stream>>>(batch, cnt, NN);
    k_zero4<<<nb(NG * 512 / 4), 256, 0, stream>>>((float4*)pooled, NG * 512 / 4);

    // layer 1: two ChebConvs concatenated into h1 [NN x 128]
    run_conv(stream, x, norm_f, src, dst, p11, 64, h1, 128, 0, B1, B2);
    k_bias_relu<<<nb(NN * 64), 256, 0, stream>>>(h1, b11, 128, 0, 64, NN);
    run_conv(stream, x, norm_r, dst, src, p12, 64, h1, 128, 64, B1, B2);
    k_bias_relu<<<nb(NN * 64), 256, 0, stream>>>(h1, b12, 128, 64, 64, NN);

    // layer 2 forward -> pooled[:, 0:256]
    run_conv(stream, h1, norm_f, src, dst, p21, 256, out2, 256, 0, B1, B2);
    k_bias_relu<<<nb(NN * 256), 256, 0, stream>>>(out2, b21, 256, 0, 256, NN);
    k_pool_add<<<nb(NN * 64), 256, 0, stream>>>(out2, batch, pooled, 256, 0, NN);

    // layer 2 reverse -> pooled[:, 256:512]  (out2 reused)
    run_conv(stream, h1, norm_r, dst, src, p22, 256, out2, 256, 0, B1, B2);
    k_bias_relu<<<nb(NN * 256), 256, 0, stream>>>(out2, b22, 256, 0, 256, NN);
    k_pool_add<<<nb(NN * 64), 256, 0, stream>>>(out2, batch, pooled, 256, 256, NN);

    // mean-pool fold (1/cnt) + FC + log_softmax
    k_fc_lsm<<<nb(NG), 256, 0, stream>>>(pooled, cnt, fcw, fcb, out);
}
